// EGConvModel_18811956757051
// MI455X (gfx1250) — compile-verified, matrix operand order checked by reading, not run on hardware
//
#include <hip/hip_runtime.h>
#include <hip/hip_bf16.h>

// ---------- types for WMMA ----------
typedef __attribute__((ext_vector_type(16))) __bf16 v16bf;
typedef __attribute__((ext_vector_type(8)))  float  v8f;

union B16x16 { uint4 q[2]; v16bf v; };

// ---------- small device helpers ----------
__device__ __forceinline__ unsigned short f2bf(float f) {
    unsigned u = __float_as_uint(f);
    unsigned r = u + 0x7fffu + ((u >> 16) & 1u);   // round-to-nearest-even
    return (unsigned short)(r >> 16);
}
// monotone float <-> uint mapping for atomicMax on floats
__device__ __forceinline__ unsigned encf(float f) {
    unsigned u = __float_as_uint(f);
    return (u & 0x80000000u) ? ~u : (u | 0x80000000u);
}
__device__ __forceinline__ float decf(unsigned u) {
    unsigned v = (u & 0x80000000u) ? (u & 0x7fffffffu) : ~u;
    return __uint_as_float(v);
}

// ---------- generic utility kernels ----------
__global__ void k_zero_f(float* p, size_t n) {
    size_t i = (size_t)blockIdx.x * blockDim.x + threadIdx.x;
    if (i < n) p[i] = 0.0f;
}
__global__ void k_fill_f(float* p, size_t n, float v) {
    size_t i = (size_t)blockIdx.x * blockDim.x + threadIdx.x;
    if (i < n) p[i] = v;
}

// ---------- scalar dense: Y[r,c] = sum_k X[r,k]*W[k,c] (+bias) ----------
__global__ void k_dense(const float* __restrict__ X, const float* __restrict__ W,
                        const float* __restrict__ bias, float* __restrict__ Y,
                        int rows, int K, int Cout) {
    long long tid = (long long)blockIdx.x * blockDim.x + threadIdx.x;
    int r = (int)(tid / Cout);
    int c = (int)(tid % Cout);
    if (r >= rows) return;
    float acc = bias ? bias[c] : 0.0f;
    const float* xr = X + (size_t)r * K;
    for (int k = 0; k < K; ++k) acc += xr[k] * W[(size_t)k * Cout + c];
    Y[(size_t)r * Cout + c] = acc;
}

// ---------- BN statistics (two-pass, biased variance like reference) ----------
__global__ void k_colsum(const float* __restrict__ arr, int rows, int C, int rpb,
                         float* __restrict__ sums, float* __restrict__ sumsq) {
    int c  = threadIdx.x;              // blockDim.x == C
    int r0 = blockIdx.x * rpb;
    int r1 = r0 + rpb; if (r1 > rows) r1 = rows;
    float s = 0.0f, q = 0.0f;
    for (int r = r0; r < r1; ++r) {
        float v = arr[(size_t)r * C + c];
        s += v; q += v * v;
    }
    atomicAdd(&sums[c], s);
    atomicAdd(&sumsq[c], q);
}
__global__ void k_bn_finalize(const float* sums, const float* sumsq, float cnt, int C,
                              float* mean, float* rstd) {
    int c = blockIdx.x * blockDim.x + threadIdx.x;
    if (c >= C) return;
    float mu  = sums[c] / cnt;
    float var = sumsq[c] / cnt - mu * mu;
    mean[c] = mu;
    rstd[c] = rsqrtf(var + 1e-5f);
}

// ---------- BN + ReLU (+optional residual), writes fp32 h and bf16 copy ----------
__global__ void k_bn_relu_update(const float* __restrict__ y, const float* __restrict__ gw,
                                 const float* __restrict__ bw, const float* __restrict__ mean,
                                 const float* __restrict__ rstd, float* __restrict__ h,
                                 unsigned short* __restrict__ hbf, size_t total, int residual) {
    size_t i = (size_t)blockIdx.x * blockDim.x + threadIdx.x;
    if (i >= total) return;
    int c = (int)(i & 127);   // C == 128
    float v = gw[c] * (y[i] - mean[c]) * rstd[c] + bw[c];
    v = fmaxf(v, 0.0f);
    float hv = residual ? (h[i] + v) : v;
    h[i]  = hv;
    hbf[i] = f2bf(hv);
}
__global__ void k_bn_relu_inplace(float* __restrict__ y, const float* __restrict__ gw,
                                  const float* __restrict__ bw, const float* __restrict__ mean,
                                  const float* __restrict__ rstd, int total, int cmask) {
    int i = blockIdx.x * blockDim.x + threadIdx.x;
    if (i >= total) return;
    int c = i & cmask;
    float v = gw[c] * (y[i] - mean[c]) * rstd[c] + bw[c];
    y[i] = fmaxf(v, 0.0f);
}

// ---------- pack weights into WMMA B-fragment lane layout (bf16) ----------
// Unified per-layer layout of 10 column tiles (tiles 0..3 = bases_w [128,64],
// tiles 4..9 = comb_w [128,96]).
// Bp index = ((tile*4 + s)*32 + lane)*16 + j ; PER_L = 10*4*32*16 = 20480.
__global__ void k_pack_weights(const float* __restrict__ bases_w,
                               const float* __restrict__ comb_w,
                               unsigned short* __restrict__ packed) {
    const int PER_L = 20480;
    int t = blockIdx.x * blockDim.x + threadIdx.x;
    if (t >= 4 * PER_L) return;
    int l = t / PER_L, e = t - l * PER_L;
    int j    = e & 15;
    int le   = e >> 4;
    int lane = le & 31;
    int ts   = le >> 5;           // tile*4 + s
    int s    = ts & 3;
    int tile = ts >> 2;           // 0..9
    const float* W; int ld, col;
    if (tile < 4) { W = bases_w + (size_t)l * 128 * 64; ld = 64; col = tile * 16 + (lane & 15); }
    else          { W = comb_w  + (size_t)l * 128 * 96; ld = 96; col = (tile - 4) * 16 + (lane & 15); }
    int koff = (lane < 16) ? 0 : 8;
    int k    = s * 32 + ((j < 8) ? (koff + j) : (16 + koff + (j - 8)));
    packed[t] = f2bf(W[(size_t)k * ld + col]);
}

// ---------- fused bf16 WMMA GEMM: bases[M,64] and comb[M,96] from A[M,128] ----------
// One wave per 16-row tile; A fragment loaded ONCE per k-step, reused across all
// 10 column tiles (4*10 = 40 v_wmma_f32_16x16x32_bf16 fully unrolled).
__global__ void k_gemm_fused(const unsigned short* __restrict__ A,
                             const unsigned short* __restrict__ Bp,
                             float* __restrict__ Cb, float* __restrict__ Cc, int M) {
    int wave = threadIdx.x >> 5;
    int lane = threadIdx.x & 31;
    int mtile = blockIdx.x * (blockDim.x >> 5) + wave;
    if (mtile >= (M >> 4)) return;                 // wave-uniform branch
    int m    = (mtile << 4) + (lane & 15);
    int koff = (lane < 16) ? 0 : 8;
    const unsigned short* arow = A + (size_t)m * 128 + koff;
    v8f acc[10] = {};
#pragma unroll
    for (int s = 0; s < 4; ++s) {
        B16x16 a;
        a.q[0] = *(const uint4*)(arow + s * 32);
        a.q[1] = *(const uint4*)(arow + s * 32 + 16);
#pragma unroll
        for (int t = 0; t < 10; ++t) {
            B16x16 b;
            const unsigned short* bp = Bp + ((size_t)(t * 4 + s) * 32 + lane) * 16;
            b.q[0] = *(const uint4*)(bp);
            b.q[1] = *(const uint4*)(bp + 8);
            acc[t] = __builtin_amdgcn_wmma_f32_16x16x32_bf16(
                false, a.v, false, b.v, (short)0, acc[t], false, false);
        }
    }
    int cl    = lane & 15;
    int mbase = (mtile << 4) + ((lane >= 16) ? 8 : 0);
#pragma unroll
    for (int t = 0; t < 4; ++t)
#pragma unroll
        for (int r = 0; r < 8; ++r)
            Cb[(size_t)(mbase + r) * 64 + t * 16 + cl] = acc[t][r];
#pragma unroll
    for (int t = 4; t < 10; ++t)
#pragma unroll
        for (int r = 0; r < 8; ++r)
            Cc[(size_t)(mbase + r) * 96 + (t - 4) * 16 + cl] = acc[t][r];
}

// ---------- degree ----------
__global__ void k_deg_edges(const int* __restrict__ dst, float* __restrict__ deg, int E_) {
    int e = blockIdx.x * blockDim.x + threadIdx.x;
    if (e >= E_) return;
    atomicAdd(&deg[dst[e]], 1.0f);
}

// ---------- aggregation init with self-loop message ----------
__global__ void k_agg_init(const float* __restrict__ bases, float* __restrict__ ssum,
                           unsigned* __restrict__ smax, size_t n) {
    size_t i = (size_t)blockIdx.x * blockDim.x + threadIdx.x;
    if (i >= n) return;
    float v = bases[i];
    ssum[i] = v;
    smax[i] = encf(v);
}

// ---------- edge scatter: sum + max (4 threads/edge, 16 floats each) ----------
__global__ void k_scatter(const int* __restrict__ src, const int* __restrict__ dst,
                          const float* __restrict__ bases, float* __restrict__ ssum,
                          unsigned* __restrict__ smax, int E_) {
    long long tid = (long long)blockIdx.x * blockDim.x + threadIdx.x;
    int e = (int)(tid >> 2);
    int part = (int)(tid & 3);
    if (e >= E_) return;
    int s = src[e], d = dst[e];
    const float* bp = bases + (size_t)s * 64 + part * 16;
    float*    sp = ssum + (size_t)d * 64 + part * 16;
    unsigned* mp = smax + (size_t)d * 64 + part * 16;
#pragma unroll
    for (int q = 0; q < 16; ++q) {
        float v = bp[q];
        atomicAdd(&sp[q], v);
        atomicMax(&mp[q], encf(v));
    }
}

// ---------- per-node einsum: out[n,h*16+f] = sum_k comb[n,h,k]*agg[n,k,f] ----------
// k = a*4 + b ; agg_a in {ssum, ssum/deg, decoded smax}
__global__ void k_einsum(const float* __restrict__ combf, const float* __restrict__ ssum,
                         const unsigned* __restrict__ smax, const float* __restrict__ deg,
                         const float* __restrict__ comb_b, const float* __restrict__ conv_bias,
                         float* __restrict__ out, int n_nodes) {
    long long tid = (long long)blockIdx.x * blockDim.x + threadIdx.x;
    int n  = (int)(tid >> 3);
    int hh = (int)(tid & 7);
    if (n >= n_nodes) return;
    float invd = 1.0f / deg[n];
    const float* cb  = combf + (size_t)n * 96 + hh * 12;
    const float* cbb = comb_b + hh * 12;
    float acc[16];
#pragma unroll
    for (int f = 0; f < 16; ++f) acc[f] = conv_bias[hh * 16 + f];
#pragma unroll
    for (int b = 0; b < 4; ++b) {
        float c0 = cb[b]     + cbb[b];       // sum
        float c1 = cb[4 + b] + cbb[4 + b];   // mean
        float c2 = cb[8 + b] + cbb[8 + b];   // max
        const float*    sp = ssum + (size_t)n * 64 + b * 16;
        const unsigned* mp = smax + (size_t)n * 64 + b * 16;
#pragma unroll
        for (int f = 0; f < 16; ++f) {
            float sv = sp[f];
            acc[f] += c0 * sv + c1 * (sv * invd) + c2 * decf(mp[f]);
        }
    }
    float* op = out + (size_t)n * 128 + hh * 16;
#pragma unroll
    for (int f = 0; f < 16; ++f) op[f] = acc[f];
}

// ---------- global mean pool ----------
__global__ void k_pool_accum(const float* __restrict__ h, const int* __restrict__ batch,
                             float* __restrict__ gsum, float* __restrict__ gcnt, size_t total) {
    size_t i = (size_t)blockIdx.x * blockDim.x + threadIdx.x;
    if (i >= total) return;
    int n = (int)(i >> 7);
    int c = (int)(i & 127);
    int bg = batch[n];
    atomicAdd(&gsum[(size_t)bg * 128 + c], h[i]);
    if (c == 0) atomicAdd(&gcnt[bg], 1.0f);
}
__global__ void k_pool_div(float* __restrict__ gsum, const float* __restrict__ gcnt, int total) {
    int i = blockIdx.x * blockDim.x + threadIdx.x;
    if (i >= total) return;
    gsum[i] /= fmaxf(gcnt[i >> 7], 1.0f);
}

// =========================== host launcher ===========================
extern "C" void kernel_launch(void* const* d_in, const int* in_sizes, int n_in,
                              void* d_out, int out_size, void* d_ws, size_t ws_size,
                              hipStream_t stream) {
    const float* x        = (const float*)d_in[0];
    const float* lin1_w   = (const float*)d_in[1];
    const float* lin1_b   = (const float*)d_in[2];
    const float* bn1_g    = (const float*)d_in[3];
    const float* bn1_b    = (const float*)d_in[4];
    const float* bases_w  = (const float*)d_in[5];
    const float* comb_w   = (const float*)d_in[6];
    const float* comb_b   = (const float*)d_in[7];
    const float* conv_bias= (const float*)d_in[8];
    const float* bn_g     = (const float*)d_in[9];
    const float* bn_b     = (const float*)d_in[10];
    const float* mlp_w1   = (const float*)d_in[11];
    const float* mbn1_g   = (const float*)d_in[12];
    const float* mbn1_b   = (const float*)d_in[13];
    const float* mlp_w2   = (const float*)d_in[14];
    const float* mbn2_g   = (const float*)d_in[15];
    const float* mbn2_b   = (const float*)d_in[16];
    const float* mlp_w3   = (const float*)d_in[17];
    const float* mlp_b3   = (const float*)d_in[18];
    const int*   edge_idx = (const int*)d_in[19];
    const int*   batch    = (const int*)d_in[20];

    const int N  = in_sizes[20];        // 100000 (multiple of 16)
    const int E_ = in_sizes[19] / 2;    // 1600000
    const int G_ = out_size / 10;       // 1000
    const int* src = edge_idx;
    const int* dst = edge_idx + E_;

    // ---- workspace carve (256B aligned regions) ----
    char* wsb = (char*)d_ws;
    size_t off = 0;
    auto take = [&](size_t bytes) -> char* {
        char* p = wsb + off;
        off += (bytes + 255) & ~(size_t)255;
        return p;
    };
    float*          h     = (float*)take((size_t)N * 128 * 4);
    unsigned short* hbf   = (unsigned short*)take((size_t)N * 128 * 2);
    float*          basesb= (float*)take((size_t)N * 64 * 4);
    float*          combfb= (float*)take((size_t)N * 96 * 4);
    float*          ssum  = (float*)take((size_t)N * 64 * 4);
    unsigned*       smax  = (unsigned*)take((size_t)N * 64 * 4);
    float*          outb  = (float*)take((size_t)N * 128 * 4);
    float*          deg   = (float*)take((size_t)N * 4);
    unsigned short* pack  = (unsigned short*)take((size_t)4 * 20480 * 2);
    float*          stats = (float*)take(512 * 4);
    float*          gsum  = (float*)take((size_t)G_ * 128 * 4);
    float*          gcnt  = (float*)take((size_t)G_ * 4);
    float*          m1    = (float*)take((size_t)G_ * 64 * 4);
    float*          m2    = (float*)take((size_t)G_ * 32 * 4);
    (void)ws_size; (void)n_in;

    float* sums  = stats;
    float* sumsq = stats + 128;
    float* mean  = stats + 256;
    float* rstd  = stats + 384;

    auto nb = [](size_t n, size_t per) { return (unsigned)((n + per - 1) / per); };
    const int T = 256;

    // ---- pack bf16 weights into WMMA B-fragment layout ----
    k_pack_weights<<<nb(4 * 20480, T), T, 0, stream>>>(bases_w, comb_w, pack);

    // ---- lin1 + BN1 + ReLU ----
    k_dense<<<nb((size_t)N * 128, T), T, 0, stream>>>(x, lin1_w, lin1_b, outb, N, 27, 128);
    k_zero_f<<<1, 256, 0, stream>>>(stats, 256);
    k_colsum<<<nb(N, 512), 128, 0, stream>>>(outb, N, 128, 512, sums, sumsq);
    k_bn_finalize<<<1, 128, 0, stream>>>(sums, sumsq, (float)N, 128, mean, rstd);
    k_bn_relu_update<<<nb((size_t)N * 128, T), T, 0, stream>>>(
        outb, bn1_g, bn1_b, mean, rstd, h, hbf, (size_t)N * 128, 0);

    // ---- degree (self loop = 1 + incoming edges) ----
    k_fill_f<<<nb(N, T), T, 0, stream>>>(deg, (size_t)N, 1.0f);
    k_deg_edges<<<nb(E_, T), T, 0, stream>>>(dst, deg, E_);

    // ---- EGConv layers ----
    const int PER_L = 20480;
    for (int l = 0; l < 4; ++l) {
        unsigned mtiles = (unsigned)(N >> 4);
        k_gemm_fused<<<nb(mtiles, 8), 256, 0, stream>>>(
            hbf, pack + (size_t)l * PER_L, basesb, combfb, N);

        k_agg_init<<<nb((size_t)N * 64, T), T, 0, stream>>>(basesb, ssum, smax, (size_t)N * 64);
        k_scatter<<<nb((size_t)E_ * 4, T), T, 0, stream>>>(src, dst, basesb, ssum, smax, E_);

        k_einsum<<<nb((size_t)N * 8, T), T, 0, stream>>>(
            combfb, ssum, smax, deg, comb_b + (size_t)l * 96,
            conv_bias + (size_t)l * 128, outb, N);

        k_zero_f<<<1, 256, 0, stream>>>(stats, 256);
        k_colsum<<<nb(N, 512), 128, 0, stream>>>(outb, N, 128, 512, sums, sumsq);
        k_bn_finalize<<<1, 128, 0, stream>>>(sums, sumsq, (float)N, 128, mean, rstd);
        k_bn_relu_update<<<nb((size_t)N * 128, T), T, 0, stream>>>(
            outb, bn_g + (size_t)l * 128, bn_b + (size_t)l * 128,
            mean, rstd, h, hbf, (size_t)N * 128, 1);
    }

    // ---- global mean pool ----
    k_zero_f<<<nb((size_t)G_ * 128, T), T, 0, stream>>>(gsum, (size_t)G_ * 128);
    k_zero_f<<<nb(G_, T), T, 0, stream>>>(gcnt, (size_t)G_);
    k_pool_accum<<<nb((size_t)N * 128, T), T, 0, stream>>>(h, batch, gsum, gcnt, (size_t)N * 128);
    k_pool_div<<<nb((size_t)G_ * 128, T), T, 0, stream>>>(gsum, gcnt, G_ * 128);

    // ---- MLP head ----
    k_dense<<<nb((size_t)G_ * 64, T), T, 0, stream>>>(gsum, mlp_w1, nullptr, m1, G_, 128, 64);
    k_zero_f<<<1, 256, 0, stream>>>(stats, 256);
    k_colsum<<<nb(G_, 256), 64, 0, stream>>>(m1, G_, 64, 256, sums, sumsq);
    k_bn_finalize<<<1, 128, 0, stream>>>(sums, sumsq, (float)G_, 64, mean, rstd);
    k_bn_relu_inplace<<<nb((size_t)G_ * 64, T), T, 0, stream>>>(
        m1, mbn1_g, mbn1_b, mean, rstd, G_ * 64, 63);

    k_dense<<<nb((size_t)G_ * 32, T), T, 0, stream>>>(m1, mlp_w2, nullptr, m2, G_, 64, 32);
    k_zero_f<<<1, 256, 0, stream>>>(stats, 256);
    k_colsum<<<nb(G_, 256), 32, 0, stream>>>(m2, G_, 32, 256, sums, sumsq);
    k_bn_finalize<<<1, 128, 0, stream>>>(sums, sumsq, (float)G_, 32, mean, rstd);
    k_bn_relu_inplace<<<nb((size_t)G_ * 32, T), T, 0, stream>>>(
        m2, mbn2_g, mbn2_b, mean, rstd, G_ * 32, 31);

    k_dense<<<nb((size_t)G_ * 10, T), T, 0, stream>>>(m2, mlp_w3, mlp_b3, (float*)d_out, G_, 32, 10);
}
